// MEGNet_2817498546586
// MI455X (gfx1250) — matrix operand order repaired and checked by reading, not compiled, not run on hardware
//
#include <hip/hip_runtime.h>
#include <math.h>

// MEGNet for MI455X (gfx1250, wave32, WMMA f16->f32 16x16x32 + async global->LDS).
// Memory-bound (~5 GB traffic, ~27 GFLOP); WMMA makes the MLP math free; edge
// gathers stream through the ASYNCcnt DMA path from f16 shadow tensors.
// Workspace requirement: ~170 MB.

typedef __attribute__((ext_vector_type(16))) _Float16 v16h;
typedef __attribute__((ext_vector_type(8)))  _Float16 v8h;
typedef __attribute__((ext_vector_type(8)))  float    v8f;

#define DF 64   // feature dim
#define HD 32   // hidden dim

// ---------------------------------------------------------------------------
// CDNA5 async global->LDS (CDNA5 ISA.3 op 98, 01_flow_sync 9.6)
// ---------------------------------------------------------------------------
__device__ __forceinline__ unsigned int lds_off32(const void* p) {
    // generic LDS pointer: addr[31:0] is the LDS byte offset (aperture mapping)
    return (unsigned int)(size_t)p;
}
__device__ __forceinline__ void async_g2l_b128(unsigned int ldsoff, const void* gptr) {
    asm volatile("global_load_async_to_lds_b128 %0, %1, off"
                 :: "v"(ldsoff), "v"((unsigned long long)(size_t)gptr)
                 : "memory");
}
__device__ __forceinline__ void wait_async0() {
    asm volatile("s_wait_asynccnt 0x0" ::: "memory");
}

// ---------------------------------------------------------------------------
// WMMA fragment helpers (CDNA5 16-bit layouts, CDNA5 ISA 7.12.2)
// ---------------------------------------------------------------------------
__device__ __forceinline__ v16h ld_a_frag(const _Float16* t, int lane, int k0, int stride) {
    int row = lane & 15;
    int klo = (lane < 16) ? 0 : 16;
    const _Float16* p = t + row * stride + k0 + klo;
    v16h a;
    *(v8h*)&a         = *(const v8h*)(p);
    *(((v8h*)&a) + 1) = *(const v8h*)(p + 8);
    return a;
}
__device__ __forceinline__ v16h ld_b_frag(const _Float16* w, int lane, int n0, int k0, int ldw) {
    int col = lane & 15;
    int kb  = (lane < 16) ? 0 : 16;
    return *(const v16h*)(w + (size_t)(n0 + col) * ldw + k0 + kb);
}
__device__ __forceinline__ v8f vzero8() {
    v8f z = {0.f, 0.f, 0.f, 0.f, 0.f, 0.f, 0.f, 0.f};
    return z;
}

// Two-layer MLP on one 16-row tile: (16xK) @ W1^T(32xK) -> +b1 -> (16x32) @ W2^T(64x32)
template <int K>
__device__ __forceinline__ void mlp_tile(const _Float16* At, _Float16* Ht,
                                         const _Float16* W1h, const _Float16* W2h,
                                         const float* b1s, int lane, v8f out[4]) {
    const int stride = K + 8;
    const int rowb = (lane < 16) ? 0 : 8;
    const int nloc = lane & 15;
    #pragma unroll
    for (int nt = 0; nt < 2; ++nt) {
        v8f acc = vzero8();
        #pragma unroll
        for (int kt = 0; kt < K / 32; ++kt) {
            v16h a = ld_a_frag(At, lane, kt * 32, stride);
            v16h b = ld_b_frag(W1h, lane, nt * 16, kt * 32, K);
            acc = __builtin_amdgcn_wmma_f32_16x16x32_f16(false, a, false, b,
                                                         (short)0, acc, false, false);
        }
        int n = nt * 16 + nloc;
        float bias = b1s[n];
        #pragma unroll
        for (int r = 0; r < 8; ++r)
            Ht[(rowb + r) * 40 + n] = (_Float16)(acc[r] + bias);
    }
    v16h a2 = ld_a_frag(Ht, lane, 0, 40);
    #pragma unroll
    for (int nt = 0; nt < 4; ++nt) {
        v16h b = ld_b_frag(W2h, lane, nt * 16, 0, 32);
        v8f acc = vzero8();
        acc = __builtin_amdgcn_wmma_f32_16x16x32_f16(false, a2, false, b,
                                                     (short)0, acc, false, false);
        out[nt] = acc;
    }
}

__device__ __forceinline__ void atomicMaxF(float* addr, float val) {
    int* ia = (int*)addr;
    int cur = __float_as_int(*addr);
    while (val > __int_as_float(cur)) {
        int prev = atomicCAS(ia, cur, __float_as_int(val));
        if (prev == cur) break;
        cur = prev;
    }
}

// ---------------------------------------------------------------------------
// Utility kernels
// ---------------------------------------------------------------------------
__global__ void k_f32_to_f16(const float* __restrict__ src, _Float16* __restrict__ dst,
                             size_t n) {
    size_t i = (size_t)blockIdx.x * blockDim.x + threadIdx.x;
    size_t stride = (size_t)gridDim.x * blockDim.x;
    for (; i < n; i += stride) dst[i] = (_Float16)src[i];
}

__global__ void k_counts_edges(const int* __restrict__ srcA, const int* __restrict__ dstA,
                               const int* __restrict__ batch, int* __restrict__ ecnt,
                               int* __restrict__ ecg, int E) {
    int e2 = blockIdx.x * blockDim.x + threadIdx.x;
    if (e2 >= 2 * E) return;
    int j = (e2 < E) ? e2 : e2 - E;
    int s, d;
    if (e2 < E) { s = srcA[j]; d = dstA[j]; } else { s = dstA[j]; d = srcA[j]; }
    atomicAdd(&ecnt[d], 1);
    atomicAdd(&ecg[batch[s]], 1);
}

__global__ void k_counts_nodes(const int* __restrict__ batch, int* __restrict__ ncg, int N) {
    int i = blockIdx.x * blockDim.x + threadIdx.x;
    if (i < N) atomicAdd(&ncg[batch[i]], 1);
}

__global__ void k_eseg(const int* __restrict__ srcA, const int* __restrict__ batch,
                       int* __restrict__ eseg, int E) {
    int i = blockIdx.x * blockDim.x + threadIdx.x;
    if (i < E) eseg[i] = batch[srcA[i]];
}

// ---------------------------------------------------------------------------
// Edge MLP: (2E) x [x_src|x_dst|ea|u] (256) -> 32 -> 64, scatter-accumulate.
// A-tile and weights staged via async global->LDS DMA from f16 shadows.
// ---------------------------------------------------------------------------
__global__ __launch_bounds__(128) void k_edge_mlp(
    const _Float16* __restrict__ xh, const _Float16* __restrict__ eah,
    const _Float16* __restrict__ uh,
    const int* __restrict__ srcA, const int* __restrict__ dstA, const int* __restrict__ batch,
    const _Float16* __restrict__ W1g, const float* __restrict__ b1,
    const _Float16* __restrict__ W2g, const float* __restrict__ b2,
    float* __restrict__ esum, float* __restrict__ egsum, float* __restrict__ epair, int E) {
    __shared__ alignas(16) _Float16 At[4][16 * 264];
    __shared__ alignas(16) _Float16 Ht[4][16 * 40];
    __shared__ alignas(16) _Float16 W1h[32 * 256];
    __shared__ alignas(16) _Float16 W2h[64 * 32];
    __shared__ float b1s[32], b2s[64];
    __shared__ int mDst[4][16], mJ[4][16], mGb[4][16], mOk[4][16];

    const int tid = threadIdx.x, wv = tid >> 5, lane = tid & 31;
    const int twoE = 2 * E;

    // async weight staging: 16B chunks, block-cooperative
    {
        unsigned int w1o = lds_off32(W1h), w2o = lds_off32(W2h);
        for (int c = tid; c < (32 * 256) / 8; c += 128)
            async_g2l_b128(w1o + c * 16, W1g + c * 8);
        for (int c = tid; c < (64 * 32) / 8; c += 128)
            async_g2l_b128(w2o + c * 16, W2g + c * 8);
    }
    if (tid < 32) b1s[tid] = b1[tid];
    if (tid < 64) b2s[tid] = b2[tid];

    // async A-tile gather: one b128 issue per 16-edge row (32 lanes x 16B = 512B row)
    const int tile0 = (blockIdx.x * 4 + wv) * 16;
    const int seg = lane >> 3;   // 0:x_src 1:x_dst 2:ea 3:u
    const int ch  = lane & 7;    // 16B chunk within 128B segment
    for (int r = 0; r < 16; ++r) {
        int e2 = tile0 + r;
        int ok = (e2 < twoE);
        int ec = ok ? e2 : 0;
        int j = (ec < E) ? ec : ec - E;
        int s, dd;
        if (ec < E) { s = srcA[j]; dd = dstA[j]; } else { s = dstA[j]; dd = srcA[j]; }
        int gb = batch[s];
        const _Float16* base =
            (seg == 0) ? xh + (size_t)s * DF :
            (seg == 1) ? xh + (size_t)dd * DF :
            (seg == 2) ? eah + (size_t)j * DF :
                         uh + (size_t)gb * DF;
        unsigned int lo = lds_off32(&At[wv][r * 264]) + seg * 128 + ch * 16;
        async_g2l_b128(lo, base + ch * 8);
        if (lane == 0) { mDst[wv][r] = dd; mJ[wv][r] = j; mGb[wv][r] = gb; mOk[wv][r] = ok; }
    }
    wait_async0();
    __syncthreads();

    v8f out[4];
    mlp_tile<256>(At[wv], Ht[wv], W1h, W2h, b1s, lane, out);

    const int rowb = (lane < 16) ? 0 : 8;
    const int nloc = lane & 15;
    for (int nt = 0; nt < 4; ++nt) {
        int n = nt * 16 + nloc;
        float bias = b2s[n];
        #pragma unroll
        for (int r = 0; r < 8; ++r) {
            int rr = rowb + r;
            if (mOk[wv][rr]) {
                float val = out[nt][r] + bias;
                atomicAdd(&esum[(size_t)mDst[wv][rr] * DF + n], val);
                atomicAdd(&egsum[(size_t)mGb[wv][rr] * DF + n], val);
                atomicAdd(&epair[(size_t)mJ[wv][rr] * DF + n], 0.5f * val);
            }
        }
    }
}

// ---------------------------------------------------------------------------
// Node MLP: N x [x|e_mean|u] (192) -> 32 -> 64; x += x_new, ngsum += x_new
// ---------------------------------------------------------------------------
__global__ __launch_bounds__(128) void k_node_mlp(
    float* __restrict__ x, const float* __restrict__ esum, const int* __restrict__ ecnt,
    const float* __restrict__ u, const int* __restrict__ batch,
    const _Float16* __restrict__ W1g, const float* __restrict__ b1,
    const _Float16* __restrict__ W2g, const float* __restrict__ b2,
    float* __restrict__ ngsum, int N) {
    __shared__ alignas(16) _Float16 At[4][16 * 200];
    __shared__ alignas(16) _Float16 Ht[4][16 * 40];
    __shared__ alignas(16) _Float16 W1h[32 * 192];
    __shared__ alignas(16) _Float16 W2h[64 * 32];
    __shared__ float b1s[32], b2s[64];
    __shared__ int mI[4][16], mB[4][16], mOk[4][16];

    const int tid = threadIdx.x, wv = tid >> 5, lane = tid & 31;

    {
        unsigned int w1o = lds_off32(W1h), w2o = lds_off32(W2h);
        for (int c = tid; c < (32 * 192) / 8; c += 128)
            async_g2l_b128(w1o + c * 16, W1g + c * 8);
        for (int c = tid; c < (64 * 32) / 8; c += 128)
            async_g2l_b128(w2o + c * 16, W2g + c * 8);
    }
    if (tid < 32) b1s[tid] = b1[tid];
    if (tid < 64) b2s[tid] = b2[tid];

    const int tile0 = (blockIdx.x * 4 + wv) * 16;
    const int c = lane * 2;
    for (int r = 0; r < 16; ++r) {
        int i = tile0 + r;
        int ok = (i < N);
        int ic = ok ? i : 0;
        int gb = batch[ic];
        float inv = 1.0f / fmaxf((float)ecnt[ic], 1.0f);
        _Float16* ar = &At[wv][r * 200];
        const float* xr = x + (size_t)ic * DF;
        const float* sr = esum + (size_t)ic * DF;
        const float* ur = u + (size_t)gb * DF;
        float2 v;
        v = *(const float2*)(xr + c); ar[c]       = (_Float16)v.x;         ar[c + 1]   = (_Float16)v.y;
        v = *(const float2*)(sr + c); ar[64 + c]  = (_Float16)(v.x * inv); ar[65 + c]  = (_Float16)(v.y * inv);
        v = *(const float2*)(ur + c); ar[128 + c] = (_Float16)v.x;         ar[129 + c] = (_Float16)v.y;
        if (lane == 0) { mI[wv][r] = ic; mB[wv][r] = gb; mOk[wv][r] = ok; }
    }
    wait_async0();
    __syncthreads();

    v8f out[4];
    mlp_tile<192>(At[wv], Ht[wv], W1h, W2h, b1s, lane, out);

    const int rowb = (lane < 16) ? 0 : 8;
    const int nloc = lane & 15;
    for (int nt = 0; nt < 4; ++nt) {
        int n = nt * 16 + nloc;
        float bias = b2s[n];
        #pragma unroll
        for (int r = 0; r < 8; ++r) {
            int rr = rowb + r;
            if (mOk[wv][rr]) {
                float val = out[nt][r] + bias;
                atomicAdd(&ngsum[(size_t)mB[wv][rr] * DF + n], val);
                size_t idx = (size_t)mI[wv][rr] * DF + n;
                x[idx] = x[idx] + val;   // residual, one writer per element
            }
        }
    }
}

// ---------------------------------------------------------------------------
// Global MLP: B x [mean_e|mean_n|u] (192) -> 32 -> 64; u += u_new. One block.
// ---------------------------------------------------------------------------
__global__ __launch_bounds__(128) void k_global_mlp(
    float* __restrict__ u, const float* __restrict__ egsum, const int* __restrict__ ecg,
    const float* __restrict__ ngsum, const int* __restrict__ ncg,
    const _Float16* __restrict__ W1g, const float* __restrict__ b1,
    const _Float16* __restrict__ W2g, const float* __restrict__ b2, int B) {
    __shared__ alignas(16) _Float16 At[4][16 * 200];
    __shared__ alignas(16) _Float16 Ht[4][16 * 40];
    __shared__ alignas(16) _Float16 W1h[32 * 192];
    __shared__ alignas(16) _Float16 W2h[64 * 32];
    __shared__ float b1s[32], b2s[64];
    __shared__ int mOk[4][16];

    const int tid = threadIdx.x, wv = tid >> 5, lane = tid & 31;

    {
        unsigned int w1o = lds_off32(W1h), w2o = lds_off32(W2h);
        for (int c = tid; c < (32 * 192) / 8; c += 128)
            async_g2l_b128(w1o + c * 16, W1g + c * 8);
        for (int c = tid; c < (64 * 32) / 8; c += 128)
            async_g2l_b128(w2o + c * 16, W2g + c * 8);
    }
    if (tid < 32) b1s[tid] = b1[tid];
    if (tid < 64) b2s[tid] = b2[tid];

    const int tile0 = wv * 16;
    const int c = lane * 2;
    for (int r = 0; r < 16; ++r) {
        int b = tile0 + r;
        int ok = (b < B);
        int bc = ok ? b : 0;
        float einv = 1.0f / fmaxf((float)ecg[bc], 1.0f);
        float ninv = 1.0f / fmaxf((float)ncg[bc], 1.0f);
        _Float16* ar = &At[wv][r * 200];
        const float* er = egsum + (size_t)bc * DF;
        const float* nr = ngsum + (size_t)bc * DF;
        const float* ur = u + (size_t)bc * DF;
        float2 v;
        v = *(const float2*)(er + c); ar[c]       = (_Float16)(v.x * einv); ar[c + 1]   = (_Float16)(v.y * einv);
        v = *(const float2*)(nr + c); ar[64 + c]  = (_Float16)(v.x * ninv); ar[65 + c]  = (_Float16)(v.y * ninv);
        v = *(const float2*)(ur + c); ar[128 + c] = (_Float16)v.x;          ar[129 + c] = (_Float16)v.y;
        if (lane == 0) mOk[wv][r] = ok;
    }
    wait_async0();
    __syncthreads();

    v8f out[4];
    mlp_tile<192>(At[wv], Ht[wv], W1h, W2h, b1s, lane, out);

    const int rowb = (lane < 16) ? 0 : 8;
    const int nloc = lane & 15;
    for (int nt = 0; nt < 4; ++nt) {
        int n = nt * 16 + nloc;
        float bias = b2s[n];
        #pragma unroll
        for (int r = 0; r < 8; ++r) {
            int rr = rowb + r;
            if (mOk[wv][rr]) {
                size_t idx = (size_t)(tile0 + rr) * DF + n;
                u[idx] = u[idx] + (out[nt][r] + bias);   // residual
            }
        }
    }
}

// ---------------------------------------------------------------------------
// Set2Set (3 steps): LSTM + segment softmax attention
// ---------------------------------------------------------------------------
__global__ void k_lstm_gates(const float* __restrict__ h, const float* __restrict__ r,
                             const float* __restrict__ Wih, const float* __restrict__ Whh,
                             const float* __restrict__ bih, const float* __restrict__ bhh,
                             float* __restrict__ gates, int B) {
    int idx = blockIdx.x * blockDim.x + threadIdx.x;
    if (idx >= B * 256) return;
    int b = idx >> 8, col = idx & 255;
    const float* wi = Wih + (size_t)col * 128;
    const float* wh = Whh + (size_t)col * 64;
    const float* hb = h + (size_t)b * 64;
    const float* rb = r + (size_t)b * 64;
    float g = bih[col] + bhh[col];
    for (int k = 0; k < 64; ++k)
        g += hb[k] * wi[k] + rb[k] * wi[64 + k] + hb[k] * wh[k];
    gates[idx] = g;
}

__global__ void k_lstm_apply(const float* __restrict__ gates, float* __restrict__ h,
                             float* __restrict__ c, float* __restrict__ r,
                             float* __restrict__ emax, float* __restrict__ denom, int B) {
    int idx = blockIdx.x * blockDim.x + threadIdx.x;
    if (idx >= B * 64) return;
    int b = idx >> 6, j = idx & 63;
    const float* g = gates + (size_t)b * 256;
    float ig = 1.f / (1.f + expf(-g[j]));
    float fg = 1.f / (1.f + expf(-g[64 + j]));
    float gg = tanhf(g[128 + j]);
    float og = 1.f / (1.f + expf(-g[192 + j]));
    float cn = fg * c[idx] + ig * gg;
    c[idx] = cn;
    h[idx] = og * tanhf(cn);
    r[idx] = 0.f;
    if (j == 0) { emax[b] = -3.0e38f; denom[b] = 0.f; }
}

__global__ void k_attn_logit(const float* __restrict__ feat, const int* __restrict__ seg,
                             const float* __restrict__ h, float* __restrict__ logit,
                             float* __restrict__ emax, int M) {
    int i = blockIdx.x * blockDim.x + threadIdx.x;
    if (i >= M) return;
    int b = seg[i];
    const float* f = feat + (size_t)i * 64;
    const float* q = h + (size_t)b * 64;
    float e = 0.f;
    for (int k = 0; k < 64; ++k) e += f[k] * q[k];
    logit[i] = e;
    atomicMaxF(&emax[b], e);
}

__global__ void k_attn_norm(float* __restrict__ logit, const int* __restrict__ seg,
                            const float* __restrict__ emax, float* __restrict__ denom, int M) {
    int i = blockIdx.x * blockDim.x + threadIdx.x;
    if (i >= M) return;
    int b = seg[i];
    float ee = expf(logit[i] - emax[b]);
    logit[i] = ee;
    atomicAdd(&denom[b], ee);
}

__global__ void k_attn_r(const float* __restrict__ logit, const int* __restrict__ seg,
                         const float* __restrict__ denom, const float* __restrict__ feat,
                         float* __restrict__ r, int M) {
    int i = blockIdx.x * blockDim.x + threadIdx.x;
    if (i >= M) return;
    int b = seg[i];
    float a = logit[i] / denom[b];
    const float* f = feat + (size_t)i * 64;
    float* rb = r + (size_t)b * 64;
    for (int k = 0; k < 64; ++k) atomicAdd(&rb[k], a * f[k]);
}

// ---------------------------------------------------------------------------
// Dense head: [hn|rn|he|re|u] (320) -> 32 -> 16 -> 1
// ---------------------------------------------------------------------------
__global__ void k_head(const float* __restrict__ hn, const float* __restrict__ rn,
                       const float* __restrict__ he, const float* __restrict__ re,
                       const float* __restrict__ u,
                       const float* __restrict__ dW1, const float* __restrict__ db1,
                       const float* __restrict__ dW2, const float* __restrict__ db2,
                       const float* __restrict__ oW, const float* __restrict__ ob,
                       float* __restrict__ out, int B) {
    int b = threadIdx.x;
    if (b >= B) return;
    const float* s0 = hn + (size_t)b * 64;
    const float* s1 = rn + (size_t)b * 64;
    const float* s2 = he + (size_t)b * 64;
    const float* s3 = re + (size_t)b * 64;
    const float* s4 = u + (size_t)b * 64;
    float t1[32];
    for (int n = 0; n < 32; ++n) {
        const float* w = dW1 + (size_t)n * 320;
        float s = db1[n];
        for (int k = 0; k < 64; ++k)
            s += s0[k] * w[k] + s1[k] * w[64 + k] + s2[k] * w[128 + k]
               + s3[k] * w[192 + k] + s4[k] * w[256 + k];
        t1[n] = s;
    }
    float t2[16];
    for (int n = 0; n < 16; ++n) {
        const float* w = dW2 + (size_t)n * 32;
        float s = db2[n];
        for (int k = 0; k < 32; ++k) s += t1[k] * w[k];
        t2[n] = s;
    }
    float o = ob[0];
    for (int k = 0; k < 16; ++k) o += t2[k] * oW[k];
    out[b] = o;
}

// ---------------------------------------------------------------------------
// Host driver
// ---------------------------------------------------------------------------
extern "C" void kernel_launch(void* const* d_in, const int* in_sizes, int n_in,
                              void* d_out, int out_size, void* d_ws, size_t ws_size,
                              hipStream_t stream) {
    (void)n_in; (void)out_size; (void)ws_size;
    const float* node_features = (const float*)d_in[0];
    const int*   edge_index    = (const int*)d_in[1];
    const float* edge_features = (const float*)d_in[2];
    const float* global_feat   = (const float*)d_in[3];
    const int*   batch         = (const int*)d_in[4];
    const float* eW1 = (const float*)d_in[5];  const float* eb1 = (const float*)d_in[6];
    const float* eW2 = (const float*)d_in[7];  const float* eb2 = (const float*)d_in[8];
    const float* nW1 = (const float*)d_in[9];  const float* nb1 = (const float*)d_in[10];
    const float* nW2 = (const float*)d_in[11]; const float* nb2 = (const float*)d_in[12];
    const float* gW1 = (const float*)d_in[13]; const float* gb1 = (const float*)d_in[14];
    const float* gW2 = (const float*)d_in[15]; const float* gb2 = (const float*)d_in[16];
    const float* sn_Wih = (const float*)d_in[17]; const float* sn_Whh = (const float*)d_in[18];
    const float* sn_bih = (const float*)d_in[19]; const float* sn_bhh = (const float*)d_in[20];
    const float* se_Wih = (const float*)d_in[21]; const float* se_Whh = (const float*)d_in[22];
    const float* se_bih = (const float*)d_in[23]; const float* se_bhh = (const float*)d_in[24];
    const float* dW1 = (const float*)d_in[25]; const float* db1 = (const float*)d_in[26];
    const float* dW2 = (const float*)d_in[27]; const float* db2 = (const float*)d_in[28];
    const float* oW  = (const float*)d_in[29]; const float* ob  = (const float*)d_in[30];

    const int N = in_sizes[0] / DF;
    const int E = in_sizes[1] / 2;
    const int B = in_sizes[3] / DF;
    const int* srcA = edge_index;
    const int* dstA = edge_index + E;

    // workspace bump allocator (256B aligned)
    char* wp = (char*)d_ws;
    auto alloc = [&](size_t bytes) -> void* {
        void* p = (void*)wp;
        wp += (bytes + 255) & ~(size_t)255;
        return p;
    };
    float*     x     = (float*)alloc((size_t)N * DF * 4);
    float*     eaA   = (float*)alloc((size_t)E * DF * 4);
    float*     eaB   = (float*)alloc((size_t)E * DF * 4);
    float*     esum  = (float*)alloc((size_t)N * DF * 4);
    int*       ecnt  = (int*)alloc((size_t)N * 4);
    float*     egsum = (float*)alloc((size_t)B * DF * 4);
    float*     ngsum = (float*)alloc((size_t)B * DF * 4);
    int*       ecg   = (int*)alloc((size_t)B * 4);
    int*       ncg   = (int*)alloc((size_t)B * 4);
    float*     ubuf  = (float*)alloc((size_t)B * DF * 4);
    int*       eseg  = (int*)alloc((size_t)E * 4);
    float*     att   = (float*)alloc((size_t)(E > N ? E : N) * 4);
    float*     gates = (float*)alloc((size_t)B * 256 * 4);
    float*     s2s   = (float*)alloc((size_t)6 * B * DF * 4);  // hn,cn,rn,he,ce,re
    float* hn = s2s, *cn = s2s + B * DF, *rn = s2s + 2 * B * DF;
    float* he = s2s + 3 * B * DF, *ce = s2s + 4 * B * DF, *re = s2s + 5 * B * DF;
    float*     emax  = (float*)alloc((size_t)B * 4);
    float*     denom = (float*)alloc((size_t)B * 4);
    // f16 shadows + f16 weights
    _Float16*  xh    = (_Float16*)alloc((size_t)N * DF * 2);
    _Float16*  eah   = (_Float16*)alloc((size_t)E * DF * 2);
    _Float16*  uh    = (_Float16*)alloc((size_t)B * DF * 2);
    _Float16*  eW1h  = (_Float16*)alloc((size_t)3 * 32 * 256 * 2);
    _Float16*  eW2h  = (_Float16*)alloc((size_t)3 * 64 * 32 * 2);
    _Float16*  nW1h  = (_Float16*)alloc((size_t)3 * 32 * 192 * 2);
    _Float16*  nW2h  = (_Float16*)alloc((size_t)3 * 64 * 32 * 2);
    _Float16*  gW1h  = (_Float16*)alloc((size_t)3 * 32 * 192 * 2);
    _Float16*  gW2h  = (_Float16*)alloc((size_t)3 * 64 * 32 * 2);

    hipMemcpyAsync(x, node_features, (size_t)N * DF * 4, hipMemcpyDeviceToDevice, stream);
    hipMemcpyAsync(eaA, edge_features, (size_t)E * DF * 4, hipMemcpyDeviceToDevice, stream);
    hipMemcpyAsync(ubuf, global_feat, (size_t)B * DF * 4, hipMemcpyDeviceToDevice, stream);
    hipMemsetAsync(ecnt, 0, (size_t)N * 4, stream);
    hipMemsetAsync(ecg, 0, (size_t)B * 4, stream);
    hipMemsetAsync(ncg, 0, (size_t)B * 4, stream);

    k_counts_edges<<<(2 * E + 255) / 256, 256, 0, stream>>>(srcA, dstA, batch, ecnt, ecg, E);
    k_counts_nodes<<<(N + 255) / 256, 256, 0, stream>>>(batch, ncg, N);
    k_eseg<<<(E + 255) / 256, 256, 0, stream>>>(srcA, batch, eseg, E);

    // convert all MLP weights to f16 once
    k_f32_to_f16<<<64, 256, 0, stream>>>(eW1, eW1h, (size_t)3 * 32 * 256);
    k_f32_to_f16<<<16, 256, 0, stream>>>(eW2, eW2h, (size_t)3 * 64 * 32);
    k_f32_to_f16<<<48, 256, 0, stream>>>(nW1, nW1h, (size_t)3 * 32 * 192);
    k_f32_to_f16<<<16, 256, 0, stream>>>(nW2, nW2h, (size_t)3 * 64 * 32);
    k_f32_to_f16<<<48, 256, 0, stream>>>(gW1, gW1h, (size_t)3 * 32 * 192);
    k_f32_to_f16<<<16, 256, 0, stream>>>(gW2, gW2h, (size_t)3 * 64 * 32);

    float* ea_cur = eaA;
    float* ea_nxt = eaB;
    for (int blk = 0; blk < 3; ++blk) {
        // refresh f16 shadows of x / ea / u for the async gather path
        k_f32_to_f16<<<1024, 256, 0, stream>>>(x, xh, (size_t)N * DF);
        k_f32_to_f16<<<2048, 256, 0, stream>>>(ea_cur, eah, (size_t)E * DF);
        k_f32_to_f16<<<4, 256, 0, stream>>>(ubuf, uh, (size_t)B * DF);

        hipMemsetAsync(esum, 0, (size_t)N * DF * 4, stream);
        hipMemsetAsync(egsum, 0, (size_t)B * DF * 4, stream);
        hipMemsetAsync(ngsum, 0, (size_t)B * DF * 4, stream);
        // e_out accumulator initialized with ea (residual); edge kernel adds 0.5*e_new twice
        hipMemcpyAsync(ea_nxt, ea_cur, (size_t)E * DF * 4, hipMemcpyDeviceToDevice, stream);

        k_edge_mlp<<<(2 * E + 63) / 64, 128, 0, stream>>>(
            xh, eah, uh, srcA, dstA, batch,
            eW1h + (size_t)blk * 32 * 256, eb1 + (size_t)blk * 32,
            eW2h + (size_t)blk * 64 * 32, eb2 + (size_t)blk * 64,
            esum, egsum, ea_nxt, E);
        k_node_mlp<<<(N + 63) / 64, 128, 0, stream>>>(
            x, esum, ecnt, ubuf, batch,
            nW1h + (size_t)blk * 32 * 192, nb1 + (size_t)blk * 32,
            nW2h + (size_t)blk * 64 * 32, nb2 + (size_t)blk * 64,
            ngsum, N);
        k_global_mlp<<<1, 128, 0, stream>>>(
            ubuf, egsum, ecg, ngsum, ncg,
            gW1h + (size_t)blk * 32 * 192, gb1 + (size_t)blk * 32,
            gW2h + (size_t)blk * 64 * 32, gb2 + (size_t)blk * 64, B);

        float* t = ea_cur; ea_cur = ea_nxt; ea_nxt = t;
    }

    // Set2Set over nodes, then edges
    hipMemsetAsync(s2s, 0, (size_t)6 * B * DF * 4, stream);
    for (int t = 0; t < 3; ++t) {
        k_lstm_gates<<<(B * 256 + 255) / 256, 256, 0, stream>>>(hn, rn, sn_Wih, sn_Whh,
                                                                sn_bih, sn_bhh, gates, B);
        k_lstm_apply<<<(B * 64 + 255) / 256, 256, 0, stream>>>(gates, hn, cn, rn, emax, denom, B);
        k_attn_logit<<<(N + 255) / 256, 256, 0, stream>>>(x, batch, hn, att, emax, N);
        k_attn_norm<<<(N + 255) / 256, 256, 0, stream>>>(att, batch, emax, denom, N);
        k_attn_r<<<(N + 255) / 256, 256, 0, stream>>>(att, batch, denom, x, rn, N);
    }
    for (int t = 0; t < 3; ++t) {
        k_lstm_gates<<<(B * 256 + 255) / 256, 256, 0, stream>>>(he, re, se_Wih, se_Whh,
                                                                se_bih, se_bhh, gates, B);
        k_lstm_apply<<<(B * 64 + 255) / 256, 256, 0, stream>>>(gates, he, ce, re, emax, denom, B);
        k_attn_logit<<<(E + 255) / 256, 256, 0, stream>>>(ea_cur, eseg, he, att, emax, E);
        k_attn_norm<<<(E + 255) / 256, 256, 0, stream>>>(att, eseg, emax, denom, E);
        k_attn_r<<<(E + 255) / 256, 256, 0, stream>>>(att, eseg, denom, ea_cur, re, E);
    }

    k_head<<<1, (B + 31) / 32 * 32, 0, stream>>>(hn, rn, he, re, ubuf,
                                                 dW1, db1, dW2, db2, oW, ob,
                                                 (float*)d_out, B);
}
